// VINN_56332791054744
// MI455X (gfx1250) — matrix operand order repaired and checked by reading, not compiled, not run on hardware
//
#include <hip/hip_runtime.h>
#include <cstdint>
#include <cmath>

// ---------------------------------------------------------------------------
// VINN top-k kernel for MI455X (gfx1250, wave32).
//   d2 = |q|^2 + |r|^2 - 2 q.r   via  v_wmma_f32_16x16x32_bf16
//   reps streamed HBM -> LDS with global_load_async_to_lds_b128 (double buffer)
//   per-thread top-k scan (all 256 threads), one block-end merge
//   final candidates rescored exactly in fp32 before softmax.
// ---------------------------------------------------------------------------

typedef __attribute__((ext_vector_type(16))) __bf16 v16bf;
typedef __attribute__((ext_vector_type(8)))  float  v8f;

#define TPB    256      // 8 wave32
#define BQ     32       // batch of queries
#define KDIM   128      // feature dim
#define ROWS   128      // rep rows per block-iteration (16 per wave)
#define QPAD   132      // padded LDS row stride (floats) -> conflict-light
#define NPART  16       // per-block per-query partial top-k
#define NACT   7        // action dim

// branch-light sorted insert into an ascending register-resident top-K list
template <int K>
__device__ __forceinline__ void topk_insert(float (&td)[K], int (&ti)[K],
                                            float d, int gi) {
  if (d >= td[K - 1]) return;
  float nd[K]; int ni[K];
#pragma unroll
  for (int j = K - 1; j >= 0; --j) {
    if (td[j] <= d)                    { nd[j] = td[j];     ni[j] = ti[j]; }
    else if (j > 0 && td[j - 1] > d)   { nd[j] = td[j - 1]; ni[j] = ti[j - 1]; }
    else                               { nd[j] = d;         ni[j] = gi; }
  }
#pragma unroll
  for (int j = 0; j < K; ++j) { td[j] = nd[j]; ti[j] = ni[j]; }
}

// Async-copy one 128x128 fp32 tile of representations into LDS (padded rows).
// 16 bytes per op, 256 threads -> 16 ops/thread, all tracked on ASYNCcnt.
__device__ __forceinline__ void async_tile_load(const float* __restrict__ reps,
                                                int N, int tileBase,
                                                float* buf, int tid) {
#pragma unroll
  for (int i = 0; i < (ROWS * KDIM / 4) / TPB; ++i) {
    int lin = i * TPB + tid;            // 0..4095 float4 slots
    int row = lin >> 5;                 // 32 float4 per row
    int c4  = lin & 31;
    int gr  = tileBase + row;
    if (gr >= N) gr = N - 1;            // clamp; filtered later by index check
    uint64_t gaddr = (uint64_t)(uintptr_t)(reps + (size_t)gr * KDIM + c4 * 4);
    // low 32 bits of a flat shared-aperture address == LDS byte offset
    uint32_t laddr = (uint32_t)(uintptr_t)(&buf[row * QPAD + c4 * 4]);
    asm volatile("global_load_async_to_lds_b128 %0, %1, off"
                 :: "v"(laddr), "v"(gaddr) : "memory");
  }
}

__global__ void __launch_bounds__(TPB)
vinn_partial_topk(const float* __restrict__ qrep,   // [32,128]
                  const float* __restrict__ reps,   // [N,128]
                  int N, int numTiles,
                  float* __restrict__ pd,           // [nb,32,16]
                  int*   __restrict__ pi) {
  extern __shared__ float smem[];
  float* repbuf0 = smem;                              // ROWS*QPAD
  float* repbuf1 = repbuf0 + ROWS * QPAD;             // ROWS*QPAD
  float* qbuf    = repbuf1 + ROWS * QPAD;             // BQ*QPAD
  float* distbuf = qbuf + BQ * QPAD;                  // BQ*QPAD
  float* q2buf   = distbuf + BQ * QPAD;               // BQ

  const int tid  = threadIdx.x;
  const int lane = tid & 31;
  const int wv   = tid >> 5;        // wave id 0..7
  const int h    = lane >> 4;       // half-wave
  const int m    = lane & 15;       // row/col-in-tile

  // ---- stage queries into LDS (fp32, padded) ----
  for (int i = tid; i < BQ * (KDIM / 4); i += TPB) {
    int row = i >> 5, c4 = i & 31;
    float4 v = reinterpret_cast<const float4*>(qrep)[row * (KDIM / 4) + c4];
    *reinterpret_cast<float4*>(&qbuf[row * QPAD + c4 * 4]) = v;
  }
  __syncthreads();

  if (tid < BQ) {
    float s0 = 0.f, s1 = 0.f, s2 = 0.f, s3 = 0.f;
    for (int j = 0; j < KDIM; j += 4) {
      float a = qbuf[tid * QPAD + j + 0], b = qbuf[tid * QPAD + j + 1];
      float c = qbuf[tid * QPAD + j + 2], d = qbuf[tid * QPAD + j + 3];
      s0 += a * a; s1 += b * b; s2 += c * c; s3 += d * d;
    }
    q2buf[tid] = (s0 + s1) + (s2 + s3);
  }
  __syncthreads();

  // ---- build bf16 A fragments (16x32, contiguous k = 16h + e per lane half), once per wave ----
  v16bf afrag[2][4];
#pragma unroll
  for (int qt = 0; qt < 2; ++qt)
#pragma unroll
    for (int kc = 0; kc < 4; ++kc)
#pragma unroll
      for (int e = 0; e < 16; ++e) {
        int K = kc * 32 + 16 * h + e;
        afrag[qt][kc][e] = (__bf16)qbuf[(qt * 16 + m) * QPAD + K];
      }
  float q2a[8], q2b[8];
#pragma unroll
  for (int j = 0; j < 8; ++j) { q2a[j] = q2buf[j + 8 * h]; q2b[j] = q2buf[16 + j + 8 * h]; }

  // per-thread top-16: thread handles query (tid&31), column chunk (tid>>5)
  const int sq = tid & 31;          // scan query
  const int sc = (tid >> 5) * 16;   // scan column base within tile
  float td[NPART]; int ti[NPART];
#pragma unroll
  for (int j = 0; j < NPART; ++j) { td[j] = INFINITY; ti[j] = -1; }

  // ---- main grid-stride loop over 128-row tiles, double-buffered async ----
  int tile = blockIdx.x;
  if (tile < numTiles) async_tile_load(reps, N, tile * ROWS, repbuf0, tid);
  int cur = 0;
  for (; tile < numTiles; tile += gridDim.x) {
    asm volatile("s_wait_asynccnt 0x0" ::: "memory");
    __syncthreads();                                  // current tile resident
    int nxt = tile + gridDim.x;
    float* buf = cur ? repbuf1 : repbuf0;
    if (nxt < numTiles)
      async_tile_load(reps, N, nxt * ROWS, cur ? repbuf0 : repbuf1, tid);

    // -- build bf16 B fragments (32x16 layout) + |r|^2 for this wave's 16 rows
    v16bf bfrag[4];
    float r0 = 0.f, r1 = 0.f, r2p = 0.f, r3 = 0.f;    // 4-way ILP for |r|^2
    const float* rrow = &buf[(wv * 16 + m) * QPAD];
#pragma unroll
    for (int kc = 0; kc < 4; ++kc) {
      int k0 = kc * 32 + 16 * h;
#pragma unroll
      for (int e4 = 0; e4 < 4; ++e4) {
        float4 v = *reinterpret_cast<const float4*>(&rrow[k0 + e4 * 4]);
        bfrag[kc][e4 * 4 + 0] = (__bf16)v.x; r0 += v.x * v.x;
        bfrag[kc][e4 * 4 + 1] = (__bf16)v.y; r1 += v.y * v.y;
        bfrag[kc][e4 * 4 + 2] = (__bf16)v.z; r2p += v.z * v.z;
        bfrag[kc][e4 * 4 + 3] = (__bf16)v.w; r3 += v.w * v.w;
      }
    }
    float r2acc = (r0 + r1) + (r2p + r3);
    float r2 = r2acc + __shfl_xor(r2acc, 16, 32);     // combine half-waves

    // -- cross products: two 16(q) x 16(r) accumulators over 4 K-chunks
    v8f acc0 = {0.f, 0.f, 0.f, 0.f, 0.f, 0.f, 0.f, 0.f};
    v8f acc1 = {0.f, 0.f, 0.f, 0.f, 0.f, 0.f, 0.f, 0.f};
#pragma unroll
    for (int kc = 0; kc < 4; ++kc) {
      acc0 = __builtin_amdgcn_wmma_f32_16x16x32_bf16(false, afrag[0][kc],
                                                     false, bfrag[kc],
                                                     (short)0, acc0, false, false);
      acc1 = __builtin_amdgcn_wmma_f32_16x16x32_bf16(false, afrag[1][kc],
                                                     false, bfrag[kc],
                                                     (short)0, acc1, false, false);
    }

    // -- distances into the 32 x 128 LDS panel (C layout: VGPR j -> row j+8h)
#pragma unroll
    for (int j = 0; j < 8; ++j) {
      float d0 = q2a[j] + r2 - 2.f * acc0[j]; d0 = d0 > 0.f ? sqrtf(d0) : 0.f;
      float d1 = q2b[j] + r2 - 2.f * acc1[j]; d1 = d1 > 0.f ? sqrtf(d1) : 0.f;
      distbuf[(j + 8 * h) * QPAD + wv * 16 + m]      = d0;
      distbuf[(16 + j + 8 * h) * QPAD + wv * 16 + m] = d1;
    }
    __syncthreads();                                  // panel complete

    // -- parallel scan: every thread filters 16 columns for its query
    {
      int gbase = tile * ROWS + sc;
      for (int c = 0; c < 16; ++c) {
        float d = distbuf[sq * QPAD + sc + c];
        int gi = gbase + c;
        if (gi < N) topk_insert<NPART>(td, ti, d, gi);
      }
    }
    __syncthreads();                                  // panel + buf reusable
    cur ^= 1;
  }

  // ---- block-end merge: 8 per-thread lists per query -> one top-16 ----
  // rep buffers are dead now; reuse as 256x16 (dist,int) staging (32 KB)
  float* md = repbuf0;
  int*   mi = reinterpret_cast<int*>(repbuf0 + TPB * NPART);
#pragma unroll
  for (int j = 0; j < NPART; ++j) {
    md[tid * NPART + j] = td[j];
    mi[tid * NPART + j] = ti[j];
  }
  __syncthreads();

  if (tid < BQ) {
    float fd[NPART]; int fi[NPART];
#pragma unroll
    for (int j = 0; j < NPART; ++j) { fd[j] = INFINITY; fi[j] = -1; }
    for (int c = 0; c < 8; ++c) {          // the 8 chunk-lists for this query
      int src = (c * 32 + tid) * NPART;
      for (int j = 0; j < NPART; ++j) {
        int gi = mi[src + j];
        if (gi >= 0) topk_insert<NPART>(fd, fi, md[src + j], gi);
      }
    }
    int base = (blockIdx.x * BQ + tid) * NPART;
#pragma unroll
    for (int j = 0; j < NPART; ++j) { pd[base + j] = fd[j]; pi[base + j] = fi[j]; }
  }
}

__global__ void __launch_bounds__(TPB)
vinn_reduce(const float* __restrict__ qrep,
            const float* __restrict__ reps,
            const float* __restrict__ actions,
            const float* __restrict__ pd, const int* __restrict__ pi,
            int nb, int N, float* __restrict__ out) {
  __shared__ float ld[TPB * NPART];
  __shared__ int   li[TPB * NPART];
  __shared__ float cd[32]; __shared__ int ci[32]; __shared__ float ce[32];

  const int q = blockIdx.x, tid = threadIdx.x;

  float td[NPART]; int ti[NPART];
#pragma unroll
  for (int j = 0; j < NPART; ++j) { td[j] = INFINITY; ti[j] = -1; }

  for (int i = tid; i < nb * NPART; i += TPB) {
    int b = i >> 4, j = i & 15;
    int idx = (b * BQ + q) * NPART + j;
    float d = pd[idx]; int gi = pi[idx];
    if (gi >= 0) topk_insert<NPART>(td, ti, d, gi);
  }
#pragma unroll
  for (int j = 0; j < NPART; ++j) { ld[tid * NPART + j] = td[j]; li[tid * NPART + j] = ti[j]; }
  __syncthreads();

  if (tid == 0) {            // merge to top-32 candidates (slack for rescore)
    float mdl[32]; int mil[32];
#pragma unroll
    for (int j = 0; j < 32; ++j) { mdl[j] = INFINITY; mil[j] = -1; }
    for (int i = 0; i < TPB * NPART; ++i)
      if (li[i] >= 0) topk_insert<32>(mdl, mil, ld[i], li[i]);
#pragma unroll
    for (int j = 0; j < 32; ++j) { cd[j] = mdl[j]; ci[j] = mil[j]; }
  }
  __syncthreads();

  if (tid < 32) {            // exact fp32 rescore, one candidate per lane
    int gi = ci[tid];
    float e = INFINITY;
    if (gi >= 0 && gi < N) {
      float s = 0.f;
      for (int j = 0; j < KDIM; ++j) {
        float diff = qrep[q * KDIM + j] - reps[(size_t)gi * KDIM + j];
        s += diff * diff;
      }
      e = sqrtf(s);
    }
    ce[tid] = e;
  }
  __syncthreads();

  if (tid == 0) {            // exact top-16, softmax(-d), weighted action sum
    float fd[NPART]; int fi[NPART];
#pragma unroll
    for (int j = 0; j < NPART; ++j) { fd[j] = INFINITY; fi[j] = -1; }
    for (int c = 0; c < 32; ++c)
      if (ci[c] >= 0) topk_insert<NPART>(fd, fi, ce[c], ci[c]);

    float mn = fd[0];                        // ascending -> min distance first
    float wsum = 0.f, pred[NACT];
#pragma unroll
    for (int a = 0; a < NACT; ++a) pred[a] = 0.f;
    for (int j = 0; j < NPART; ++j) {
      if (fi[j] < 0) continue;
      float w = __expf(mn - fd[j]);
      wsum += w;
      for (int a = 0; a < NACT; ++a)
        pred[a] += w * actions[(size_t)fi[j] * NACT + a];
    }
    float inv = 1.f / wsum;
    for (int a = 0; a < NACT; ++a) out[q * NACT + a] = pred[a] * inv;
  }
}

extern "C" void kernel_launch(void* const* d_in, const int* in_sizes, int n_in,
                              void* d_out, int out_size, void* d_ws, size_t ws_size,
                              hipStream_t stream) {
  const float* qrep    = (const float*)d_in[0];   // [32,128]
  const float* reps    = (const float*)d_in[1];   // [N,128]
  const float* actions = (const float*)d_in[2];   // [N,7]
  (void)d_in[3]; (void)n_in; (void)out_size;      // k == 16, compile-time

  const int N        = in_sizes[1] / KDIM;
  const int numTiles = (N + ROWS - 1) / ROWS;

  // per-block partials: 32 queries * 16 entries * (4B dist + 4B idx) = 4096 B
  int nb = (int)(ws_size / (size_t)(BQ * NPART * 8));
  if (nb > 1024) nb = 1024;
  if (nb > numTiles) nb = numTiles;
  if (nb < 1) nb = 1;

  float* pd = (float*)d_ws;
  int*   pi = (int*)(pd + (size_t)nb * BQ * NPART);

  constexpr size_t SMEM =
      (size_t)(2 * ROWS * QPAD + 2 * BQ * QPAD + BQ) * sizeof(float); // ~169 KB
  hipFuncSetAttribute((const void*)vinn_partial_topk,
                      hipFuncAttributeMaxDynamicSharedMemorySize, (int)SMEM);

  vinn_partial_topk<<<nb, TPB, SMEM, stream>>>(qrep, reps, N, numTiles, pd, pi);
  vinn_reduce<<<BQ, TPB, 0, stream>>>(qrep, reps, actions, pd, pi, nb, N,
                                      (float*)d_out);
}
